// CrossAttention_63299228008650
// MI455X (gfx1250) — matrix operand order repaired and checked by reading, not compiled, not run on hardware
//
#include <hip/hip_runtime.h>

typedef _Float16 f16;
typedef _Float16 v8h  __attribute__((ext_vector_type(8)));
typedef _Float16 v16h __attribute__((ext_vector_type(16)));
typedef float    v8f  __attribute__((ext_vector_type(8)));
typedef unsigned int u32x4 __attribute__((ext_vector_type(4)));
typedef int          i32x4 __attribute__((ext_vector_type(4)));
typedef int          i32x8 __attribute__((ext_vector_type(8)));

union Frag { v16h v; v8h h[2]; };

#define WMMA_F16(a,b,c) __builtin_amdgcn_wmma_f32_16x16x32_f16(false,(a),false,(b),(short)0,(c),false,false)

constexpr int B_ = 4, S_ = 3072, SKV_ = 1036, SKVP_ = 1056;
constexpr int D_ = 1024, DC_ = 768, H_ = 8, DH_ = 128;

// ------------------------------------------------------------ TDM support ---
#if __has_builtin(__builtin_amdgcn_tensor_load_to_lds) && __has_builtin(__builtin_amdgcn_s_wait_tensorcnt)
#define HAVE_TDM 1
#endif

#ifdef HAVE_TDM
// 2D tile load: tile_y rows of tile_x f16 elements, row stride = stride_elems.
// Rows/cols beyond tensor_x/tensor_y are zero-filled by the TDM (OOB rule),
// packed per CDNA5 ISA D# groups 0/1 (data_size=2B, type=2).
__device__ __forceinline__ void tdm_load_2d_f16(unsigned lds_off, const f16* g,
                                                unsigned tile_x, unsigned tile_y,
                                                unsigned tensor_x, unsigned tensor_y,
                                                unsigned stride_elems)
{
    unsigned long long ga = (unsigned long long)(const void*)g;
    u32x4 g0 = { 1u,                                   // count=1, user mode, no gather
                 lds_off,                              // lds_addr [63:32]
                 (unsigned)(ga & 0xffffffffu),         // global_addr[31:0]
                 (unsigned)((ga >> 32) & 0x01ffffffu) | (2u << 30) };  // addr[56:32] | type=2
    i32x8 g1 = { (int)0x00010000,                      // workgroup_mask=0, data_size=1 (2B)
                 (int)((tensor_x & 0xffffu) << 16),    // tensor_dim0[15:0]
                 (int)((tensor_x >> 16) | ((tensor_y & 0xffffu) << 16)),   // dim0 hi | dim1 lo
                 (int)((tensor_y >> 16) | (tile_x << 16)),                 // dim1 hi | tile_dim0
                 (int)(tile_y & 0xffffu),              // tile_dim1, tile_dim2=0
                 (int)stride_elems,                    // tensor_dim0_stride[31:0]
                 0, 0 };
    i32x4 z4 = { 0, 0, 0, 0 };
#if __has_include(<hip/amd_detail/amd_gfx1250_TDM.h>)
    i32x8 z8 = { 0, 0, 0, 0, 0, 0, 0, 0 };
    __builtin_amdgcn_tensor_load_to_lds(g0, g1, z4, z4, z8, 0);   // 6-arg toolchain
#else
    __builtin_amdgcn_tensor_load_to_lds(g0, g1, z4, z4, 0);       // 5-arg toolchain (ROCm 7.2)
#endif
}
#endif

// ---------------------------------------------------------------- convert ---
__global__ void cvt_f32_to_f16(const float* __restrict__ in, f16* __restrict__ out, long n) {
    long i = (long)blockIdx.x * blockDim.x + threadIdx.x;
    long stride = (long)gridDim.x * blockDim.x;
    for (; i < n; i += stride) {
        if (i + stride < n) __builtin_prefetch(in + i + stride, 0, 1);
        out[i] = (f16)in[i];
    }
}

// ------------------------------------------------------------------- GEMM ---
// C[m,n] = sum_k A[m,k] * W[n,k] + bias[n]   (A @ W.T + b)
// 256 threads = 8 waves, block tile 128x128, k-step 64 (16 WMMA / barrier),
// double-buffered LDS slabs, TDM prefetch of the next slab when available.
// MODE 0: f16 row-major; MODE 1: f16 transposed (V^T); MODE 2: f32 row-major.
template<int MODE>
__global__ __launch_bounds__(256)
void gemm128(const f16* __restrict__ A, const f16* __restrict__ W,
             const float* __restrict__ bias, void* __restrict__ Cout,
             int M_real, int M_store, int K, int ldc,
             long strideA, long strideC)
{
    __shared__ __align__(16) f16 sA[2][128 * 64];
    __shared__ __align__(16) f16 sB[2][128 * 64];
    const int tid = threadIdx.x;
    const int wave = tid >> 5, lane = tid & 31;
    const int row16 = lane & 15, hi = lane >> 4;
    const int wm = wave & 3, wn = wave >> 2;
    const int m0 = blockIdx.y * 128, n0 = blockIdx.x * 128;
    A += (long)blockIdx.z * strideA;

    v8f acc[2][4] = {};
    const int nsteps = K >> 6;
    const unsigned remM = (M_real > m0) ? (unsigned)(M_real - m0) : 0u;  // TDM zero-fills rows >= remM

#ifdef HAVE_TDM
    if (wave == 0) {   // prologue: slab 0
        tdm_load_2d_f16((unsigned)(size_t)&sA[0][0], A + (long)m0 * K, 64, 128, 64, remM, K);
        tdm_load_2d_f16((unsigned)(size_t)&sB[0][0], W + (long)n0 * K, 64, 128, 64, 128, K);
    }
#endif

    for (int ks = 0; ks < nsteps; ++ks) {
        const int k0 = ks * 64;
        const int buf = ks & 1;

#ifdef HAVE_TDM
        if (wave == 0) __builtin_amdgcn_s_wait_tensorcnt(0);  // slab `ks` arrived
        __syncthreads();                                       // publish + retire old buf reads
        if (wave == 0 && ks + 1 < nsteps) {                    // prefetch slab ks+1
            tdm_load_2d_f16((unsigned)(size_t)&sA[buf ^ 1][0],
                            A + (long)m0 * K + k0 + 64, 64, 128, 64, remM, K);
            tdm_load_2d_f16((unsigned)(size_t)&sB[buf ^ 1][0],
                            W + (long)n0 * K + k0 + 64, 64, 128, 64, 128, K);
        }
#else
        __syncthreads();
        #pragma unroll
        for (int i = 0; i < 4; ++i) {
            int e = tid * 8 + i * 2048;
            int r = e >> 6, c = e & 63;
            v8h va = {};
            if (m0 + r < M_real) va = *(const v8h*)(A + (long)(m0 + r) * K + k0 + c);
            *(v8h*)&sA[buf][e] = va;
            *(v8h*)&sB[buf][e] = *(const v8h*)(W + (long)(n0 + r) * K + k0 + c);
        }
        __syncthreads();
#endif

        #pragma unroll
        for (int kk = 0; kk < 2; ++kk) {
            Frag a[2], b[4];
            #pragma unroll
            for (int mi = 0; mi < 2; ++mi) {
                int mr = wm * 32 + mi * 16 + row16;
                a[mi].h[0] = *(const v8h*)&sA[buf][mr * 64 + kk * 32 + hi * 8];
                a[mi].h[1] = *(const v8h*)&sA[buf][mr * 64 + kk * 32 + 16 + hi * 8];
            }
            #pragma unroll
            for (int ni = 0; ni < 4; ++ni) {
                int nr = wn * 64 + ni * 16 + row16;
                b[ni].h[0] = *(const v8h*)&sB[buf][nr * 64 + kk * 32 + hi * 8];
                b[ni].h[1] = *(const v8h*)&sB[buf][nr * 64 + kk * 32 + 16 + hi * 8];
            }
            #pragma unroll
            for (int mi = 0; mi < 2; ++mi)
                #pragma unroll
                for (int ni = 0; ni < 4; ++ni)
                    acc[mi][ni] = WMMA_F16(a[mi].v, b[ni].v, acc[mi][ni]);
        }
    }

    long cb = (long)blockIdx.z * strideC;
    #pragma unroll
    for (int mi = 0; mi < 2; ++mi)
        #pragma unroll
        for (int ni = 0; ni < 4; ++ni)
            #pragma unroll
            for (int r = 0; r < 8; ++r) {
                int m = m0 + wm * 32 + mi * 16 + r + 8 * hi;
                int n = n0 + wn * 64 + ni * 16 + row16;
                float v = acc[mi][ni][r] + bias[n];
                if (MODE == 0) { if (m < M_store) ((f16*)Cout)[cb + (long)m * ldc + n] = (f16)v; }
                else if (MODE == 1) { if (m < M_store) ((f16*)Cout)[cb + (long)n * ldc + m] = (f16)v; }
                else { if (m < M_real) ((float*)Cout)[cb + (long)m * ldc + n] = v; }
            }
}

// -------------------------------------------------------- flash attention ---
// grid (S/128, H, B), block 256 (8 waves). Each wave: 16 q rows, d_head=128.
// K (32x128) and V^T (128x32) tiles staged once per block in LDS
// (TDM double-buffered when available, cooperative loads otherwise).
__global__ __launch_bounds__(256)
void attn128(const f16* __restrict__ Qh, const f16* __restrict__ Kh,
             const f16* __restrict__ VT, f16* __restrict__ Oh)
{
    __shared__ __align__(16) f16 sK[2][32 * 128];
    __shared__ __align__(16) f16 sV[2][128 * 32];
    __shared__ __align__(16) f16 sP[8 * 16 * 32];
    const int tid = threadIdx.x, wave = tid >> 5, lane = tid & 31;
    const int row16 = lane & 15, hi = lane >> 4;
    const int b = blockIdx.z, h = blockIdx.y;
    const int q0 = blockIdx.x * 128 + wave * 16;

    const f16* Qb = Qh + ((long)b * S_ + q0) * D_ + h * DH_;
    const f16* Kb = Kh + (long)b * SKVP_ * D_ + h * DH_;
    const f16* Vb = VT + ((long)b * D_ + h * DH_) * SKVP_;
    f16*       Ob = Oh + ((long)b * S_ + q0) * D_ + h * DH_;

    // Q fragments held in registers for the whole kv loop
    Frag qa[4];
    #pragma unroll
    for (int d = 0; d < 4; ++d) {
        const f16* p = Qb + (long)row16 * D_ + d * 32;
        qa[d].h[0] = *(const v8h*)(p + hi * 8);
        qa[d].h[1] = *(const v8h*)(p + 16 + hi * 8);
    }

    v8f o[8] = {};
    float mrow[8], lrow[8];
    #pragma unroll
    for (int r = 0; r < 8; ++r) { mrow[r] = -1e30f; lrow[r] = 0.f; }
    const float scale = 0.08838834764831845f;  // 1/sqrt(128)
    f16* sPw = sP + wave * 512;

    const int nsteps = (SKV_ + 31) / 32;   // 33 steps, kv padded to 1056

#ifdef HAVE_TDM
    if (wave == 0) {   // prologue: tile 0
        tdm_load_2d_f16((unsigned)(size_t)&sK[0][0], Kb, 128, 32, 128, 32, D_);
        tdm_load_2d_f16((unsigned)(size_t)&sV[0][0], Vb, 32, 128, 32, 128, SKVP_);
    }
#endif

    for (int it = 0; it < nsteps; ++it) {
        const int kv0 = it * 32;
        const int buf = it & 1;

#ifdef HAVE_TDM
        if (wave == 0) __builtin_amdgcn_s_wait_tensorcnt(0);  // tile `it` arrived
        __syncthreads();
        if (wave == 0 && it + 1 < nsteps) {                    // prefetch tile it+1
            const int nkv = kv0 + 32;
            tdm_load_2d_f16((unsigned)(size_t)&sK[buf ^ 1][0], Kb + (long)nkv * D_,
                            128, 32, 128, 32, D_);
            tdm_load_2d_f16((unsigned)(size_t)&sV[buf ^ 1][0], Vb + nkv,
                            32, 128, 32, 128, SKVP_);
        }
#else
        __syncthreads();
        #pragma unroll
        for (int i = 0; i < 2; ++i) {
            int e = tid * 8 + i * 2048;
            int rK = e >> 7, cK = e & 127;
            *(v8h*)&sK[buf][e] = *(const v8h*)(Kb + (long)(kv0 + rK) * D_ + cK);
            int rV = e >> 5, cV = e & 31;
            *(v8h*)&sV[buf][e] = *(const v8h*)(Vb + (long)rV * SKVP_ + kv0 + cV);
        }
        __syncthreads();
#endif

        const f16* kbuf = &sK[buf][0];
        const f16* vbuf = &sV[buf][0];

        // scores: Q(16x128) x K_tile(32x128)^T -> two 16x16 f32 tiles
        v8f s0 = {}, s1 = {};
        #pragma unroll
        for (int d = 0; d < 4; ++d) {
            Frag kb0, kb1;
            kb0.h[0] = *(const v8h*)&kbuf[row16 * 128 + d * 32 + hi * 8];
            kb0.h[1] = *(const v8h*)&kbuf[row16 * 128 + d * 32 + 16 + hi * 8];
            kb1.h[0] = *(const v8h*)&kbuf[(16 + row16) * 128 + d * 32 + hi * 8];
            kb1.h[1] = *(const v8h*)&kbuf[(16 + row16) * 128 + d * 32 + 16 + hi * 8];
            s0 = WMMA_F16(qa[d].v, kb0.v, s0);
            s1 = WMMA_F16(qa[d].v, kb1.v, s1);
        }

        // online softmax (rows split across wave halves; shfl_xor {8,4,2,1} stays in-half)
        const int c0 = kv0 + row16, c1 = c0 + 16;
        #pragma unroll
        for (int r = 0; r < 8; ++r) {
            float x0 = (c0 < SKV_) ? s0[r] * scale : -1e30f;
            float x1 = (c1 < SKV_) ? s1[r] * scale : -1e30f;
            float mx = fmaxf(x0, x1);
            #pragma unroll
            for (int off = 8; off >= 1; off >>= 1) mx = fmaxf(mx, __shfl_xor(mx, off, 32));
            float mnew = fmaxf(mrow[r], mx);
            float p0 = __expf(x0 - mnew);
            float p1 = __expf(x1 - mnew);
            float rs = p0 + p1;
            #pragma unroll
            for (int off = 8; off >= 1; off >>= 1) rs += __shfl_xor(rs, off, 32);
            float alpha = __expf(mrow[r] - mnew);
            mrow[r] = mnew;
            lrow[r] = lrow[r] * alpha + rs;
            #pragma unroll
            for (int j = 0; j < 8; ++j) o[j][r] *= alpha;
            sPw[(r + 8 * hi) * 32 + row16]      = (f16)p0;
            sPw[(r + 8 * hi) * 32 + 16 + row16] = (f16)p1;
        }

        // P (16x32) as one A-fragment, then O += P x V_tile (8 WMMA over d_head)
        Frag pa;
        pa.h[0] = *(const v8h*)&sPw[row16 * 32 + hi * 8];
        pa.h[1] = *(const v8h*)&sPw[row16 * 32 + 16 + hi * 8];
        #pragma unroll
        for (int j = 0; j < 8; ++j) {
            Frag vb;
            vb.h[0] = *(const v8h*)&vbuf[(j * 16 + row16) * 32 + hi * 8];
            vb.h[1] = *(const v8h*)&vbuf[(j * 16 + row16) * 32 + 16 + hi * 8];
            o[j] = WMMA_F16(pa.v, vb.v, o[j]);
        }
    }

    #pragma unroll
    for (int r = 0; r < 8; ++r) {
        float inv = 1.f / lrow[r];
        int m = r + 8 * hi;
        #pragma unroll
        for (int j = 0; j < 8; ++j)
            Ob[(long)m * D_ + j * 16 + row16] = (f16)(o[j][r] * inv);
    }
}

// ------------------------------------------------------------------- host ---
extern "C" void kernel_launch(void* const* d_in, const int* in_sizes, int n_in,
                              void* d_out, int out_size, void* d_ws, size_t ws_size,
                              hipStream_t stream)
{
    const float* x  = (const float*)d_in[0];
    const float* y  = (const float*)d_in[1];
    const float* Wq = (const float*)d_in[2];
    const float* bq = (const float*)d_in[3];
    const float* Wk = (const float*)d_in[4];
    const float* bk = (const float*)d_in[5];
    const float* Wv = (const float*)d_in[6];
    const float* bv = (const float*)d_in[7];
    const float* Wo = (const float*)d_in[8];
    const float* bo = (const float*)d_in[9];

    char* ws = (char*)d_ws;
    size_t off = 0;
    auto alloc = [&](size_t bytes) {
        void* p = ws + off;
        off = (off + bytes + 255) & ~(size_t)255;
        return p;
    };
    f16* xh = (f16*)alloc((size_t)B_ * S_ * D_ * 2);
    f16* yh = (f16*)alloc((size_t)B_ * SKV_ * DC_ * 2);
    f16* wq = (f16*)alloc((size_t)D_ * D_ * 2);
    f16* wk = (f16*)alloc((size_t)D_ * DC_ * 2);
    f16* wv = (f16*)alloc((size_t)D_ * DC_ * 2);
    f16* wo = (f16*)alloc((size_t)D_ * D_ * 2);
    f16* qh = (f16*)alloc((size_t)B_ * S_ * D_ * 2);       // Q   [B,S,1024]
    f16* kh = (f16*)alloc((size_t)B_ * SKVP_ * D_ * 2);    // K   [B,1056,1024]
    f16* vt = (f16*)alloc((size_t)B_ * D_ * SKVP_ * 2);    // V^T [B,1024,1056]
    f16* oh = (f16*)alloc((size_t)B_ * S_ * D_ * 2);       // attn out [B,S,1024]
    (void)ws_size; (void)in_sizes; (void)n_in; (void)out_size;

    auto cvt = [&](const float* src, f16* dst, long n) {
        int blocks = (int)((n + 1023) / 1024);
        cvt_f32_to_f16<<<blocks, 256, 0, stream>>>(src, dst, n);
    };
    cvt(x,  xh, (long)B_ * S_ * D_);
    cvt(y,  yh, (long)B_ * SKV_ * DC_);
    cvt(Wq, wq, (long)D_ * D_);
    cvt(Wk, wk, (long)D_ * DC_);
    cvt(Wv, wv, (long)D_ * DC_);
    cvt(Wo, wo, (long)D_ * D_);

    gemm128<0><<<dim3(D_ / 128, S_ / 128, B_), 256, 0, stream>>>(
        xh, wq, bq, qh, S_, S_, D_, D_, (long)S_ * D_, (long)S_ * D_);
    gemm128<0><<<dim3(D_ / 128, (SKVP_ + 127) / 128, B_), 256, 0, stream>>>(
        yh, wk, bk, kh, SKV_, SKVP_, DC_, D_, (long)SKV_ * DC_, (long)SKVP_ * D_);
    gemm128<1><<<dim3(D_ / 128, (SKVP_ + 127) / 128, B_), 256, 0, stream>>>(
        yh, wv, bv, vt, SKV_, SKVP_, DC_, SKVP_, (long)SKV_ * DC_, (long)D_ * SKVP_);
    attn128<<<dim3(S_ / 128, H_, B_), 256, 0, stream>>>(qh, kh, vt, oh);
    gemm128<2><<<dim3(D_ / 128, S_ / 128, B_), 256, 0, stream>>>(
        oh, wo, bo, (float*)d_out, S_, S_, D_, D_, (long)S_ * D_, (long)S_ * D_);
}
